// SelfAttention_11845519802899
// MI455X (gfx1250) — matrix operand order repaired and checked by reading, not compiled, not run on hardware
//
#include <hip/hip_runtime.h>
#include <hip/hip_bf16.h>

// ---------------------------------------------------------------------------
// Self-attention (B=4, N=2048, C=1024, H=8, full-width heads) for gfx1250.
// All GEMM-like math runs on v_wmma_f32_16x16x32_bf16 with f32 accumulation.
// GEMM B-tiles are DMA'd into double-buffered LDS by the Tensor Data Mover,
// overlapped with WMMA compute and synchronized via TENSORcnt.
// ---------------------------------------------------------------------------

typedef __attribute__((ext_vector_type(16))) __bf16 v16bf;
typedef __attribute__((ext_vector_type(8)))  __bf16 v8bf;
typedef __attribute__((ext_vector_type(8)))  float  v8f;
typedef __attribute__((ext_vector_type(4)))  unsigned int u32x4;
typedef __attribute__((ext_vector_type(4)))  int i32x4;
typedef __attribute__((ext_vector_type(8)))  int i32x8;

#define B_    4
#define N_    2048
#define C_    1024
#define H_    8
#define HC_   8192            // H*C
#define ST_   24576           // 3*H*C token stride in qkv buffer
#define SCALE_ 0.08838834764831845f   // (1024/8)^-0.5

#if defined(__has_builtin)
#if __has_builtin(__builtin_amdgcn_tensor_load_to_lds) && __has_builtin(__builtin_amdgcn_s_wait_tensorcnt)
#define HAVE_TDM 1
#endif
#endif

static __device__ __forceinline__ __bf16 f2bf(float f) {
  unsigned u = __builtin_bit_cast(unsigned, f);
  unsigned r = u + 0x7FFFu + ((u >> 16) & 1u);   // round-to-nearest-even
  unsigned short h = (unsigned short)(r >> 16);
  return __builtin_bit_cast(__bf16, h);
}

static __device__ __forceinline__ v16bf cat16(v8bf lo, v8bf hi) {
  return __builtin_shufflevector(lo, hi, 0,1,2,3,4,5,6,7,8,9,10,11,12,13,14,15);
}

#ifdef HAVE_TDM
// Issue a TDM 2D tile load (bf16 elements) global -> LDS.
//   tile: tile_d1 rows x tile_d0 elements, row stride stride0 elements.
//   LDS rows padded: pad_interval_code/pad_amount_code per D# group1 encoding.
static __device__ __forceinline__ void tdm_load_2d(
    unsigned lds_off, const void* gptr,
    unsigned tensor_d0, unsigned tensor_d1,
    unsigned tile_d0, unsigned tile_d1,
    unsigned long long stride0,
    unsigned pad_interval_code, unsigned pad_amount_code, bool pad_en) {
  unsigned long long ga = (unsigned long long)gptr;
  u32x4 g0;
  g0[0] = 1u;                                           // count=1, user D#
  g0[1] = lds_off;                                      // lds_addr (bytes)
  g0[2] = (unsigned)(ga & 0xFFFFFFFFu);                 // global_addr lo
  g0[3] = (unsigned)((ga >> 32) & 0x1FFFFFFu) | (2u << 30);  // addr hi | type=2
  i32x8 g1;
  unsigned dw0 = (1u << 16)                             // data_size = 2 bytes
               | (pad_en ? (1u << 20) : 0u)             // pad_enable
               | (pad_interval_code << 22)
               | (pad_amount_code << 25);
  g1[0] = (int)dw0;
  g1[1] = (int)((tensor_d0 & 0xFFFFu) << 16);                       // td0[15:0]
  g1[2] = (int)((tensor_d0 >> 16) | ((tensor_d1 & 0xFFFFu) << 16)); // td0 hi|td1 lo
  g1[3] = (int)((tensor_d1 >> 16) | (tile_d0 << 16));               // td1 hi|tile0
  g1[4] = (int)(tile_d1 & 0xFFFFu);                                 // tile1|tile2=0
  g1[5] = (int)(unsigned)(stride0 & 0xFFFFFFFFu);
  g1[6] = (int)(unsigned)((stride0 >> 32) & 0xFFFFu);               // |stride1=0
  g1[7] = 0;
  i32x4 z4 = (i32x4){0, 0, 0, 0};
#if __clang_major__ >= 23
  i32x8 z8 = (i32x8){0, 0, 0, 0, 0, 0, 0, 0};
  __builtin_amdgcn_tensor_load_to_lds(g0, g1, z4, z4, z8, 0);
#else
  __builtin_amdgcn_tensor_load_to_lds(g0, g1, z4, z4, 0);
#endif
}
#endif

// ---------------------------------------------------------------------------
// fp32 -> bf16 conversion
// ---------------------------------------------------------------------------
__global__ __launch_bounds__(256) void cvt_f32_bf16(const float* __restrict__ in,
                                                    __bf16* __restrict__ out, int n) {
  int i = blockIdx.x * blockDim.x + threadIdx.x;
  if (i < n) out[i] = f2bf(in[i]);
}

// ---------------------------------------------------------------------------
// C = A @ B^T.  A: MxK row-major bf16.  B: NxK row-major bf16.
// 256 threads = 8 waves; 128x128 block tile; each wave owns 16 rows x 128
// cols (8 f32 accumulators).  B tile (128x32) double-buffered in LDS, filled
// by the Tensor Data Mover overlapped with WMMA compute; A fragments direct
// from global (operands are L2-resident: 192MB L2 vs <70MB bf16 operands).
// ---------------------------------------------------------------------------
template <bool OUT_BF16>
__global__ __launch_bounds__(256) void gemm_bf16_wmma(
    const __bf16* __restrict__ A, const __bf16* __restrict__ Bm,
    void* __restrict__ Cout, const float* __restrict__ bias,
    int M, int N, int K) {
  (void)M;
  __shared__ __align__(16) __bf16 Bs[2][128][40];  // 64B rows + 16B pad, x2 buf
  const int wave = threadIdx.x >> 5;
  const int lane = threadIdx.x & 31;
  const int r    = lane & 15;
  const int koff = (lane >> 4) << 4;   // A-fragment K offset (0 or 16)
  const int nb   = (lane >> 4) << 4;   // B-fragment K offset (0 or 16)
  const int n0 = blockIdx.x * 128;
  const int m0 = blockIdx.y * 128 + wave * 16;

  v8f acc[8];
#pragma unroll
  for (int t = 0; t < 8; ++t) acc[t] = (v8f){0,0,0,0,0,0,0,0};

  const __bf16* btile = Bm + (size_t)n0 * K;

#ifdef HAVE_TDM
  const unsigned bs_base = (unsigned)(size_t)(&Bs[0][0][0]); // LDS byte offset
  // Prologue: DMA tile 0 into buffer 0 and publish it.
  if (wave == 0) {
    tdm_load_2d(bs_base, btile, (unsigned)K, (unsigned)N, 32u, 128u,
                (unsigned long long)K, 3u, 3u, true);
    __builtin_amdgcn_s_wait_tensorcnt(0);
  }
  __syncthreads();

  int cur = 0;
  for (int k0 = 0; k0 < K; k0 += 32) {
    // Kick off DMA for the next tile into the free buffer (overlaps compute).
    if (wave == 0 && k0 + 32 < K)
      tdm_load_2d(bs_base + (unsigned)(cur ^ 1) * (128u * 40u * 2u),
                  btile + k0 + 32, (unsigned)K, (unsigned)N, 32u, 128u,
                  (unsigned long long)K, 3u, 3u, true);

    const __bf16* ap = A + (size_t)(m0 + r) * K + k0 + koff;
    v16bf afrag = cat16(*(const v8bf*)ap, *(const v8bf*)(ap + 8));
#pragma unroll
    for (int t = 0; t < 8; ++t) {
      const __bf16* bp = &Bs[cur][t * 16 + r][nb];
      v16bf bfrag = cat16(*(const v8bf*)bp, *(const v8bf*)(bp + 8));
      acc[t] = __builtin_amdgcn_wmma_f32_16x16x32_bf16(
          false, afrag, false, bfrag, (short)0, acc[t], false, false);
    }

    __syncthreads();                      // all waves done reading Bs[cur]
    if (wave == 0 && k0 + 32 < K)
      __builtin_amdgcn_s_wait_tensorcnt(0);
    __syncthreads();                      // publish Bs[cur^1]
    cur ^= 1;
  }
#else
  for (int k0 = 0; k0 < K; k0 += 32) {
    __syncthreads();
    { // cooperative stage: thread t -> row t/2, half-chunk (t&1)*16
      const int n  = threadIdx.x >> 1;
      const int kp = (threadIdx.x & 1) << 4;
      const __bf16* src = btile + (size_t)n * K + k0 + kp;
      *(v8bf*)&Bs[0][n][kp]     = *(const v8bf*)src;
      *(v8bf*)&Bs[0][n][kp + 8] = *(const v8bf*)(src + 8);
      if (k0 + 32 < K) __builtin_prefetch(src + 32, 0, 0);
    }
    __syncthreads();
    const __bf16* ap = A + (size_t)(m0 + r) * K + k0 + koff;
    v16bf afrag = cat16(*(const v8bf*)ap, *(const v8bf*)(ap + 8));
#pragma unroll
    for (int t = 0; t < 8; ++t) {
      const __bf16* bp = &Bs[0][t * 16 + r][nb];
      v16bf bfrag = cat16(*(const v8bf*)bp, *(const v8bf*)(bp + 8));
      acc[t] = __builtin_amdgcn_wmma_f32_16x16x32_bf16(
          false, afrag, false, bfrag, (short)0, acc[t], false, false);
    }
  }
#endif

  const int mb = (lane >> 4) << 3;
#pragma unroll
  for (int t = 0; t < 8; ++t) {
    const int col = n0 + t * 16 + r;
#pragma unroll
    for (int j = 0; j < 8; ++j) {
      const size_t row = (size_t)(m0 + mb + j);
      if (OUT_BF16) ((__bf16*)Cout)[row * N + col] = f2bf(acc[t][j]);
      else          ((float*)Cout)[row * N + col]  = acc[t][j] + bias[col];
    }
  }
}

// ---------------------------------------------------------------------------
// Flash attention, one block per (b, h, 16-row Q tile); 8 waves split the
// 1024-wide head dim (128 dims each).  Online softmax across 64 blocks of
// 32 keys.  QK^T partials reduced across waves via ds_add_f32 into LDS.
// ---------------------------------------------------------------------------
__global__ __launch_bounds__(256) void attn_fa_wmma(const __bf16* __restrict__ qkv,
                                                    __bf16* __restrict__ att_out) {
  const int blk = blockIdx.x;
  const int qt = blk & 127;          // N/16 = 128 q tiles
  const int h  = (blk >> 7) & 7;
  const int b  = blk >> 10;
  const int wave = threadIdx.x >> 5;
  const int lane = threadIdx.x & 31;
  const int r    = lane & 15;
  const int koff = (lane >> 4) << 4;
  const int nb   = (lane >> 4) << 4;
  const int mb   = (lane >> 4) << 3;
  const int r0   = qt * 16;
  const int dim0 = wave * 128;       // this wave's head-dim slice

  __shared__ __align__(16) float  S[16][33];     // scores 16x32 (f32 reduce)
  __shared__ float row_m[16], row_l[16], corr_s[16];
  __shared__ __align__(16) __bf16 P[16][40];     // exp(scores) as bf16
  __shared__ __align__(16) __bf16 Vt[8][16][40]; // per-wave V^T staging (16c x 32k)

  const __bf16* Qp = qkv + (size_t)b * N_ * ST_ + (size_t)h * C_;
  const __bf16* Kp = Qp + HC_;
  const __bf16* Vp = Qp + 2 * HC_;

  // Preload Q fragments for this wave's 128-dim slice (4 K-chunks of 32).
  v16bf a_q[4];
#pragma unroll
  for (int c = 0; c < 4; ++c) {
    const __bf16* p = Qp + (size_t)(r0 + r) * ST_ + dim0 + c * 32 + koff;
    a_q[c] = cat16(*(const v8bf*)p, *(const v8bf*)(p + 8));
  }

  if (threadIdx.x < 16) { row_m[threadIdx.x] = -__builtin_inff(); row_l[threadIdx.x] = 0.f; }

  v8f acc[8];
#pragma unroll
  for (int t = 0; t < 8; ++t) acc[t] = (v8f){0,0,0,0,0,0,0,0};

  for (int kb = 0; kb < N_ / 32; ++kb) {
    __syncthreads();
    { int i = threadIdx.x; S[i >> 5][i & 31] = 0.f; i += 256; S[i >> 5][i & 31] = 0.f; }
    __syncthreads();

    // ---- partial scores for this wave's dim slice: 2 n-tiles x 4 k-chunks
    v8f c0 = (v8f){0,0,0,0,0,0,0,0};
    v8f c1 = (v8f){0,0,0,0,0,0,0,0};
#pragma unroll
    for (int c = 0; c < 4; ++c) {
      const __bf16* k0p = Kp + (size_t)(kb * 32 + r) * ST_ + dim0 + c * 32 + nb;
      v16bf bk0 = cat16(*(const v8bf*)k0p, *(const v8bf*)(k0p + 8));
      const __bf16* k1p = k0p + (size_t)16 * ST_;
      v16bf bk1 = cat16(*(const v8bf*)k1p, *(const v8bf*)(k1p + 8));
      c0 = __builtin_amdgcn_wmma_f32_16x16x32_bf16(false, a_q[c], false, bk0, (short)0, c0, false, false);
      c1 = __builtin_amdgcn_wmma_f32_16x16x32_bf16(false, a_q[c], false, bk1, (short)0, c1, false, false);
    }
#pragma unroll
    for (int j = 0; j < 8; ++j) {
      atomicAdd(&S[mb + j][r],      c0[j]);   // ds_add_f32 cross-wave reduce
      atomicAdd(&S[mb + j][16 + r], c1[j]);
    }
    __syncthreads();

    // ---- online softmax: one thread per row
    if (threadIdx.x < 16) {
      const int row = threadIdx.x;
      float mo = row_m[row], mx = mo;
      for (int j = 0; j < 32; ++j) mx = fmaxf(mx, S[row][j] * SCALE_);
      const float corr = __expf(mo - mx);
      float l = row_l[row] * corr;
      for (int j = 0; j < 32; ++j) {
        float p = __expf(S[row][j] * SCALE_ - mx);
        l += p;
        P[row][j] = f2bf(p);
      }
      row_m[row] = mx; row_l[row] = l; corr_s[row] = corr;
    }
    __syncthreads();

    // ---- rescale accumulators by correction factor
    float cj[8];
#pragma unroll
    for (int j = 0; j < 8; ++j) cj[j] = corr_s[mb + j];
#pragma unroll
    for (int t = 0; t < 8; ++t)
#pragma unroll
      for (int j = 0; j < 8; ++j) acc[t][j] *= cj[j];

    // ---- P @ V for this wave's 128 columns
    const __bf16* pp = &P[r][koff];
    v16bf a_p = cat16(*(const v8bf*)pp, *(const v8bf*)(pp + 8));
#pragma unroll
    for (int t = 0; t < 8; ++t) {
      // stage V block transposed (16 cols x 32 keys): vectorized 16B global
      // loads (8 contiguous cols), scalar b16 scatter into LDS transpose.
#pragma unroll
      for (int it = 0; it < 2; ++it) {
        const int idx = lane + it * 32;        // 0..63
        const int key = idx >> 1;              // 0..31
        const int cg  = (idx & 1) << 3;        // col group: 0 or 8
        v8bf vv = *(const v8bf*)(Vp + (size_t)(kb * 32 + key) * ST_ + dim0 + t * 16 + cg);
#pragma unroll
        for (int e = 0; e < 8; ++e) Vt[wave][cg + e][key] = vv[e];
      }
      const __bf16* vp = &Vt[wave][r][nb];
      v16bf bv = cat16(*(const v8bf*)vp, *(const v8bf*)(vp + 8));
      acc[t] = __builtin_amdgcn_wmma_f32_16x16x32_bf16(false, a_p, false, bv, (short)0, acc[t], false, false);
    }
  }

  // ---- finalize: divide by softmax denominator, store bf16 (B,N,H*C)
  float inv[8];
#pragma unroll
  for (int j = 0; j < 8; ++j) inv[j] = 1.0f / row_l[mb + j];
#pragma unroll
  for (int t = 0; t < 8; ++t)
#pragma unroll
    for (int j = 0; j < 8; ++j) {
      const size_t row = (size_t)b * N_ + r0 + mb + j;
      att_out[row * HC_ + h * C_ + dim0 + t * 16 + r] = f2bf(acc[t][j] * inv[j]);
    }
}

// ---------------------------------------------------------------------------
// attn_mask output: broadcast (B,N) bool mask to (B,N,C) float 0/1
// ---------------------------------------------------------------------------
__global__ __launch_bounds__(256) void mask_fill(const unsigned char* __restrict__ mask,
                                                 float* __restrict__ out, int total) {
  int i = blockIdx.x * blockDim.x + threadIdx.x;
  if (i < total) out[i] = mask[i >> 10] ? 1.0f : 0.0f;   // i>>10 == token idx (C=1024)
}

// ---------------------------------------------------------------------------
extern "C" void kernel_launch(void* const* d_in, const int* in_sizes, int n_in,
                              void* d_out, int out_size, void* d_ws, size_t ws_size,
                              hipStream_t stream) {
  (void)in_sizes; (void)n_in; (void)out_size; (void)ws_size;
  const float*         x     = (const float*)d_in[0];
  const unsigned char* mask  = (const unsigned char*)d_in[1];
  const float*         Wqkv  = (const float*)d_in[2];
  const float*         Wproj = (const float*)d_in[3];
  const float*         bproj = (const float*)d_in[4];
  float* out = (float*)d_out;

  // workspace layout (bf16 scratch)
  __bf16* xb     = (__bf16*)d_ws;                          //  8,388,608
  __bf16* wqkvb  = xb     + (size_t)8192 * 1024;           // 25,165,824
  __bf16* wprojb = wqkvb  + (size_t)24576 * 1024;          //  8,388,608
  __bf16* qkvb   = wprojb + (size_t)1024 * 8192;           // 201,326,592
  __bf16* attb   = qkvb   + (size_t)8192 * 24576;          // 67,108,864

  const int nx = 8192 * 1024, nwq = 24576 * 1024, nwp = 1024 * 8192;
  cvt_f32_bf16<<<(nx  + 255) / 256, 256, 0, stream>>>(x,     xb,     nx);
  cvt_f32_bf16<<<(nwq + 255) / 256, 256, 0, stream>>>(Wqkv,  wqkvb,  nwq);
  cvt_f32_bf16<<<(nwp + 255) / 256, 256, 0, stream>>>(Wproj, wprojb, nwp);

  // qkv = x @ Wqkv^T   (M=8192, N=24576, K=1024), bf16 out
  gemm_bf16_wmma<true><<<dim3(24576 / 128, 8192 / 128), 256, 0, stream>>>(
      xb, wqkvb, (void*)qkvb, nullptr, 8192, 24576, 1024);

  // flash attention -> (B,N,H*C) bf16
  attn_fa_wmma<<<B_ * H_ * (N_ / 16), 256, 0, stream>>>(qkvb, attb);

  // out = att @ Wproj^T + bproj   (M=8192, N=1024, K=8192), f32 out
  gemm_bf16_wmma<false><<<dim3(1024 / 128, 8192 / 128), 256, 0, stream>>>(
      attb, wprojb, (void*)out, bproj, 8192, 1024, 8192);

  // second output: broadcast mask
  const int nm = 8192 * 1024;
  mask_fill<<<(nm + 255) / 256, 256, 0, stream>>>(mask, out + nm, nm);
}
